// GATv2Conv_58463094833356
// MI455X (gfx1250) — matrix operand order repaired and checked by reading, not compiled, not run on hardware
//
#include <hip/hip_runtime.h>
#include <hip/hip_bf16.h>
#include <math.h>

typedef __attribute__((ext_vector_type(16))) _Float16 v16h;
typedef __attribute__((ext_vector_type(8)))  float    v8f;

#define NEG_SLOPE 0.2f
#define EPS 1e-16f
#define IN_DIM 128
#define HC 64          // H*C
#define NHEAD 2
#define CDIM 32
#define W_ELEMS (2 * IN_DIM * HC)   // both weight matrices, f16-packed

// ---------------------------------------------------------------------------
// Kernel 0: pack lin_w / res_w into f16, pre-swizzled into the WMMA B-frag
// layout:  wsw[((w*4+kc)*4+nt)*512 + lane*16 + e] = W[(kc*32+((e>>3)<<4)+((lane>>4)<<3)+(e&7))*64 + nt*16+(lane&15)]
// so each wave's B fragment is one contiguous 32B chunk per lane.
// ---------------------------------------------------------------------------
__global__ __launch_bounds__(256)
void gat_pack_weights(const float* __restrict__ lin_w,
                      const float* __restrict__ res_w,
                      _Float16* __restrict__ wsw)
{
    int i = blockIdx.x * blockDim.x + threadIdx.x;
    if (i >= W_ELEMS) return;
    const int e    =  i        & 15;
    const int lane = (i >> 4)  & 31;
    const int nt   = (i >> 9)  & 3;
    const int kc   = (i >> 11) & 3;
    const int w    = (i >> 13) & 1;
    const float* W = w ? res_w : lin_w;
    wsw[i] = (_Float16)W[(kc * 32 + ((e >> 3) << 4) + ((lane >> 4) << 3) + (e & 7)) * HC + nt * 16 + (lane & 15)];
}

// ---------------------------------------------------------------------------
// Kernel 1: fused  h = x@lin_w   and   out = x@res_w + bias   via WMMA f16.
// One wave per 16-row tile of x; 2 weights x 4 N-tiles x 4 K-steps = 32 WMMAs.
// ---------------------------------------------------------------------------
__global__ __launch_bounds__(128)
void gat_fused_gemm(const float* __restrict__ x,
                    const _Float16* __restrict__ wsw,
                    const float* __restrict__ bias,
                    float* __restrict__ h,
                    float* __restrict__ out,
                    int N)
{
    const int wave_global = (blockIdx.x * blockDim.x + threadIdx.x) >> 5;
    const int lane = threadIdx.x & 31;
    const int base = wave_global * 16;
    if (base >= N) return;

    const int half = lane >> 4;     // 0: lanes 0-15, 1: lanes 16-31
    const int lr   = lane & 15;
    const bool full = (base + 16 <= N);   // wave-uniform fast path

    int arow = base + lr;
    if (arow >= N) arow = N - 1;    // clamp (harmless duplicate load)
    const float* xr = x + (size_t)arow * IN_DIM;

    // --- Build A fragments: 16-bit A 16x32 layout ---------------------------
    // lanes 0-15 : row=lr, elems 0-7 -> K=kc*32+0..7,  elems 8-15 -> K=+16..23
    // lanes16-31 : row=lr, elems 0-7 -> K=kc*32+8..15, elems 8-15 -> K=+24..31
    v16h afrag[4];
#pragma unroll
    for (int kc = 0; kc < 4; ++kc) {
        const int k0 = kc * 32 + half * 8;
        const int k1 = kc * 32 + 16 + half * 8;
#pragma unroll
        for (int e = 0; e < 8; ++e) {
            afrag[kc][e]     = (_Float16)xr[k0 + e];
            afrag[kc][8 + e] = (_Float16)xr[k1 + e];
        }
    }

    const v16h* __restrict__ wv = (const v16h*)wsw;   // 32B-aligned frags

#pragma unroll
    for (int w = 0; w < 2; ++w) {
        float* D = (w == 0) ? h : out;
#pragma unroll
        for (int nt = 0; nt < 4; ++nt) {
            v8f acc = {};
#pragma unroll
            for (int kc = 0; kc < 4; ++kc) {
                // pre-swizzled B frag: contiguous 16 f16 per lane
                const v16h bfrag = wv[(((w * 4 + kc) * 4) + nt) * 32 + lane];
                acc = __builtin_amdgcn_wmma_f32_16x16x32_f16(
                        /*neg_a=*/false, afrag[kc],
                        /*neg_b=*/false, bfrag,
                        /*c_mod=*/(short)0, acc,
                        /*reuse_a=*/false, /*reuse_b=*/false);
            }
            // D layout: vgpr g -> M = g + 8*half, lane lr -> N
            // Fast path: one base pointer, 8 stores at immediate offsets.
            float* Dp = D + (size_t)(base + 8 * half) * HC + nt * 16 + lr;
            if (full) {
                if (w == 1) {
                    const float badd = bias[nt * 16 + lr];
#pragma unroll
                    for (int g = 0; g < 8; ++g)
                        Dp[(size_t)g * HC] = acc[g] + badd;
                } else {
#pragma unroll
                    for (int g = 0; g < 8; ++g)
                        Dp[(size_t)g * HC] = acc[g];
                }
            } else {  // ragged last tile (uniform branch; rarely taken)
                const float badd = (w == 1) ? bias[nt * 16 + lr] : 0.0f;
#pragma unroll
                for (int g = 0; g < 8; ++g) {
                    const int row = base + g + 8 * half;
                    if (row < N)
                        Dp[(size_t)g * HC] = acc[g] + badd;
                }
            }
        }
    }
}

// ---------------------------------------------------------------------------
// Kernel 2: init m = -inf, denom = 0
// ---------------------------------------------------------------------------
__global__ void gat_init_md(float* __restrict__ m, float* __restrict__ denom,
                            long long n)
{
    long long i = (long long)blockIdx.x * blockDim.x + threadIdx.x;
    if (i < n) {
        m[i] = -INFINITY;
        denom[i] = 0.0f;
    }
}

__device__ inline void atomicMaxFloat(float* addr, float val)
{
    unsigned int* ua = (unsigned int*)addr;
    unsigned int old = __float_as_uint(*addr);
    while (__uint_as_float(old) < val) {
        unsigned int assumed = old;
        old = atomicCAS(ua, assumed, __float_as_uint(val));
        if (old == assumed) break;
    }
}

// ---------------------------------------------------------------------------
// Kernel 3: per-edge attention scores + segment max (h is L2-resident)
// ---------------------------------------------------------------------------
__global__ __launch_bounds__(256)
void gat_edge_scores(const float* __restrict__ h,
                     const long long* __restrict__ src,
                     const long long* __restrict__ dst,
                     const float* __restrict__ att,
                     float* __restrict__ ebuf,
                     float* __restrict__ mmax,
                     long long E)
{
    long long e = (long long)blockIdx.x * blockDim.x + threadIdx.x;
    if (e >= E) return;
    // prefetch next tile of the streaming index arrays (global_prefetch_b8)
    __builtin_prefetch(src + e + 4096, 0, 0);
    __builtin_prefetch(dst + e + 4096, 0, 0);
    const long long s = src[e];
    const long long d = dst[e];
    const float* hs = h + s * HC;
    const float* hd = h + d * HC;
#pragma unroll
    for (int hh = 0; hh < NHEAD; ++hh) {
        float acc = 0.0f;
#pragma unroll
        for (int c = 0; c < CDIM; c += 4) {
            const float4 a  = *(const float4*)(hs + hh * CDIM + c);
            const float4 b  = *(const float4*)(hd + hh * CDIM + c);
            const float4 at = *(const float4*)(att + hh * CDIM + c);
            float v;
            v = a.x + b.x; v = v > 0.0f ? v : NEG_SLOPE * v; acc += v * at.x;
            v = a.y + b.y; v = v > 0.0f ? v : NEG_SLOPE * v; acc += v * at.y;
            v = a.z + b.z; v = v > 0.0f ? v : NEG_SLOPE * v; acc += v * at.z;
            v = a.w + b.w; v = v > 0.0f ? v : NEG_SLOPE * v; acc += v * at.w;
        }
        ebuf[e * NHEAD + hh] = acc;
        atomicMaxFloat(&mmax[d * NHEAD + hh], acc);
    }
}

// ---------------------------------------------------------------------------
// Kernel 4: e_exp = exp(e - m[dst]); denom[dst] += e_exp   (both heads/thread)
// ---------------------------------------------------------------------------
__global__ __launch_bounds__(256)
void gat_edge_exp(const long long* __restrict__ dst,
                  const float* __restrict__ mmax,
                  float* __restrict__ ebuf,
                  float* __restrict__ denom,
                  long long E)
{
    long long e = (long long)blockIdx.x * blockDim.x + threadIdx.x;
    if (e >= E) return;
    const long long d = dst[e];
#pragma unroll
    for (int hh = 0; hh < NHEAD; ++hh) {
        const float v = __expf(ebuf[e * NHEAD + hh] - mmax[d * NHEAD + hh]);
        ebuf[e * NHEAD + hh] = v;
        atomicAdd(&denom[d * NHEAD + hh], v);
    }
}

// ---------------------------------------------------------------------------
// Kernel 5: scatter  out[dst] += h[src] * alpha.  32 lanes per edge; alpha
// computed once on lane 0 and broadcast (wave32 shuffle) to avoid 32x
// redundant ebuf/denom loads.  Native global_atomic_add_f32 accumulation.
// ---------------------------------------------------------------------------
__global__ __launch_bounds__(256)
void gat_edge_scatter(const float* __restrict__ h,
                      const long long* __restrict__ src,
                      const long long* __restrict__ dst,
                      const float* __restrict__ ebuf,
                      const float* __restrict__ denom,
                      float* __restrict__ out,
                      long long E)
{
    long long t = (long long)blockIdx.x * blockDim.x + threadIdx.x;
    const long long e = t >> 5;
    const int c = (int)(t & 31);
    if (e >= E) return;
    const long long s = src[e];
    const long long d = dst[e];
    float a0 = 0.0f, a1 = 0.0f;
    if (c == 0) {
        a0 = ebuf[e * NHEAD + 0] / (denom[d * NHEAD + 0] + EPS);
        a1 = ebuf[e * NHEAD + 1] / (denom[d * NHEAD + 1] + EPS);
    }
    a0 = __shfl(a0, 0, 32);
    a1 = __shfl(a1, 0, 32);
    atomicAdd(&out[d * HC + c],        h[s * HC + c]        * a0);
    atomicAdd(&out[d * HC + CDIM + c], h[s * HC + CDIM + c] * a1);
}

// ---------------------------------------------------------------------------
extern "C" void kernel_launch(void* const* d_in, const int* in_sizes, int n_in,
                              void* d_out, int out_size, void* d_ws, size_t ws_size,
                              hipStream_t stream)
{
    const float*     x     = (const float*)d_in[0];
    const long long* eidx  = (const long long*)d_in[1];   // int64 in reference
    const float*     lin_w = (const float*)d_in[2];
    const float*     att   = (const float*)d_in[3];
    const float*     res_w = (const float*)d_in[4];
    const float*     bias  = (const float*)d_in[5];
    float*           out   = (float*)d_out;

    const int       N = in_sizes[0] / IN_DIM;
    const long long E = in_sizes[1] / 2;
    const long long* src = eidx;
    const long long* dst = eidx + E;

    // workspace carve-out (all offsets 32B-aligned)
    float*     h     = (float*)d_ws;                 // N * 64 f32
    float*     mmax  = h + (size_t)N * HC;           // N * 2  f32
    float*     denom = mmax + (size_t)N * NHEAD;     // N * 2  f32
    float*     ebuf  = denom + (size_t)N * NHEAD;    // E * 2  f32
    _Float16*  wsw   = (_Float16*)(ebuf + (size_t)E * NHEAD);  // 16384 f16

    // 0) pack weights to f16 in WMMA B-frag layout (once per launch)
    gat_pack_weights<<<(W_ELEMS + 255) / 256, 256, 0, stream>>>(lin_w, res_w, wsw);

    // 1) fused GEMMs (writes h and out = x@res_w + bias)
    {
        const int waves = (N + 15) / 16;
        const int blocks = (waves + 3) / 4;            // 4 waves / block
        gat_fused_gemm<<<blocks, 128, 0, stream>>>(x, wsw, bias, h, out, N);
    }
    // 2) init segment max / denom
    {
        const long long n = (long long)N * NHEAD;
        gat_init_md<<<(int)((n + 255) / 256), 256, 0, stream>>>(mmax, denom, n);
    }
    // 3) edge scores + segment max
    gat_edge_scores<<<(int)((E + 255) / 256), 256, 0, stream>>>(h, src, dst, att,
                                                               ebuf, mmax, E);
    // 4) exp + segment sum
    gat_edge_exp<<<(int)((E + 255) / 256), 256, 0, stream>>>(dst, mmax, ebuf,
                                                             denom, E);
    // 5) weighted scatter into out (residual already there)
    {
        const long long th = E * 32;
        gat_edge_scatter<<<(int)((th + 255) / 256), 256, 0, stream>>>(
            h, src, dst, ebuf, denom, out, E);
    }
}
